// ParallelCoAttentionNetwork_57320633532536
// MI455X (gfx1250) — matrix operand order repaired and checked by reading, not compiled, not run on hardware
//
#include <hip/hip_runtime.h>
#include <hip/hip_bf16.h>
#include <math.h>

// ---------------- CDNA5 WMMA types ----------------
typedef __bf16 bf16_t;
typedef bf16_t v16bf __attribute__((ext_vector_type(16)));
typedef float  v8f   __attribute__((ext_vector_type(8)));
typedef unsigned int v4u __attribute__((ext_vector_type(4)));

typedef __attribute__((address_space(1))) int* gint_p;   // global int*
typedef __attribute__((address_space(3))) int* lint_p;   // LDS int*

#define WMMA_BF16(a, b, c) \
  __builtin_amdgcn_wmma_f32_16x16x32_bf16(false, (a), false, (b), (short)0, (c), false, false)

union Frag16 { v16bf v; v4u q[2]; };
union Pack8  { bf16_t h[8]; v4u q; };

// ---------------- problem dims ----------------
static constexpr int B_  = 128;
static constexpr int H_  = 512;
static constexpr int R_  = 196;
static constexpr int L_  = 1024;
static constexpr int CO_ = 256;
static constexpr int RP_ = 224;   // R padded to multiple of 32 (K-dim of H_q GEMM)

// =================================================================
// 64x32 macro-tile bf16 GEMM core: D(64x32,f32) += A[M,K] * B[K,N]
// A row-major [M][K]; B given transposed row-major Bt[N][K].
// One wave per macro tile: 8 v_wmma per K-step from 12 b128 loads
// (A frags reused x2, B frags reused x4 -> ~21 FLOP per cache byte).
//
// Fragment lane layouts (CDNA5 ISA 7.12.2, wave32):
//   A 16x32 bf16 : lane(g,i) i=M row; elems 0..7 = K(8g+0..7), 8..15 = K(16+8g+0..7)
//   B 32x16 bf16 : lane(g,i) i=N col; elems 0..7 = K(8g+0..7), 8..15 = K(16+8g+0..7)
//   D 16x16 f32  : lane(g,i) i=N col; VGPR j = row M=j+8g
// =================================================================
__device__ __forceinline__ void mm64x32(const bf16_t* __restrict__ A, int lda,
                                        const bf16_t* __restrict__ Bt, int ldb,
                                        int K, int m0, int n0, v8f acc[8]) {
  const int lane = threadIdx.x & 31;
  const int g = lane >> 4;
  const int i = lane & 15;
  const bf16_t* pa[4];
  const bf16_t* pb[2];
#pragma unroll
  for (int mt = 0; mt < 4; ++mt) pa[mt] = A + (size_t)(m0 + mt * 16 + i) * lda + 8 * g;
#pragma unroll
  for (int nt = 0; nt < 2; ++nt) pb[nt] = Bt + (size_t)(n0 + nt * 16 + i) * ldb + 8 * g;

  for (int k = 0; k < K; k += 32) {
    Frag16 a[4], b[2];
#pragma unroll
    for (int nt = 0; nt < 2; ++nt) {
      b[nt].q[0] = *(const v4u*)(pb[nt] + k);
      b[nt].q[1] = *(const v4u*)(pb[nt] + k + 16);
    }
#pragma unroll
    for (int mt = 0; mt < 4; ++mt) {
      a[mt].q[0] = *(const v4u*)(pa[mt] + k);
      a[mt].q[1] = *(const v4u*)(pa[mt] + k + 16);
    }
    if (k + 128 < K) {                 // gfx1250 global_prefetch_b8
      __builtin_prefetch(pa[0] + k + 128, 0, 3);
      __builtin_prefetch(pb[0] + k + 128, 0, 3);
    }
#pragma unroll
    for (int mt = 0; mt < 4; ++mt) {
      acc[mt * 2 + 0] = WMMA_BF16(a[mt].v, b[0].v, acc[mt * 2 + 0]);
      acc[mt * 2 + 1] = WMMA_BF16(a[mt].v, b[1].v, acc[mt * 2 + 1]);
    }
  }
}

// ---- GEMM, store D transposed bf16: OutT[n][m] (per-lane contiguous 16B) ----
__global__ void k_gemm_storeT(const bf16_t* A, int lda, size_t sA,
                              const bf16_t* Bt, int ldb, size_t sB,
                              bf16_t* OutT, int ldo, size_t sO, int K) {
  const int b = blockIdx.z;
  const int m0 = blockIdx.y * 64, n0 = blockIdx.x * 32;
  v8f acc[8] = {};
  mm64x32(A + sA * b, lda, Bt + sB * b, ldb, K, m0, n0, acc);
  const int lane = threadIdx.x & 31, g = lane >> 4, i = lane & 15;
  bf16_t* O = OutT + sO * b;
#pragma unroll
  for (int mt = 0; mt < 4; ++mt)
#pragma unroll
    for (int nt = 0; nt < 2; ++nt) {
      Pack8 o; v8f a = acc[mt * 2 + nt];
#pragma unroll
      for (int j = 0; j < 8; ++j) o.h[j] = (bf16_t)a[j];
      *(v4u*)(O + (size_t)(n0 + nt * 16 + i) * ldo + m0 + mt * 16 + 8 * g) = o.q;
    }
}

// ---- GEMM, store D normal bf16: Out[m][n] ----
__global__ void k_gemm_storeN(const bf16_t* A, int lda, size_t sA,
                              const bf16_t* Bt, int ldb, size_t sB,
                              bf16_t* Out, int ldo, size_t sO, int K) {
  const int b = blockIdx.z;
  const int m0 = blockIdx.y * 64, n0 = blockIdx.x * 32;
  v8f acc[8] = {};
  mm64x32(A + sA * b, lda, Bt + sB * b, ldb, K, m0, n0, acc);
  const int lane = threadIdx.x & 31, g = lane >> 4, i = lane & 15;
  bf16_t* O = Out + sO * b;
#pragma unroll
  for (int mt = 0; mt < 4; ++mt)
#pragma unroll
    for (int nt = 0; nt < 2; ++nt) {
      v8f a = acc[mt * 2 + nt];
#pragma unroll
      for (int j = 0; j < 8; ++j)
        O[(size_t)(m0 + mt * 16 + 8 * g + j) * ldo + n0 + nt * 16 + i] = (bf16_t)a[j];
    }
}

// ---- GEMM, store D in BOTH orientations (for C and C^T) ----
__global__ void k_gemm_storeBoth(const bf16_t* A, int lda, size_t sA,
                                 const bf16_t* Bt, int ldb, size_t sB,
                                 bf16_t* OutN, int ldn, size_t sN,
                                 bf16_t* OutT, int ldt, size_t sT, int K) {
  const int b = blockIdx.z;
  const int m0 = blockIdx.y * 64, n0 = blockIdx.x * 32;
  v8f acc[8] = {};
  mm64x32(A + sA * b, lda, Bt + sB * b, ldb, K, m0, n0, acc);
  const int lane = threadIdx.x & 31, g = lane >> 4, i = lane & 15;
  bf16_t* ON = OutN + sN * b;
  bf16_t* OT = OutT + sT * b;
#pragma unroll
  for (int mt = 0; mt < 4; ++mt)
#pragma unroll
    for (int nt = 0; nt < 2; ++nt) {
      Pack8 o; v8f a = acc[mt * 2 + nt];
#pragma unroll
      for (int j = 0; j < 8; ++j) {
        bf16_t hv = (bf16_t)a[j];
        o.h[j] = hv;
        ON[(size_t)(m0 + mt * 16 + 8 * g + j) * ldn + n0 + nt * 16 + i] = hv;
      }
      *(v4u*)(OT + (size_t)(n0 + nt * 16 + i) * ldt + m0 + mt * 16 + 8 * g) = o.q;
    }
}

// ---- GEMM fused epilogue: h = tanh(acc + Add[c][n]); sOut[n] += w[c]*h ----
// (never materializes H_v / H_q; reduces over channel rows c = M dim)
__global__ void k_gemm_tanh_reduce(const bf16_t* A, int lda, size_t sA,
                                   const bf16_t* Bt, int ldb, size_t sB,
                                   const bf16_t* Add, int ldadd, size_t sAdd,
                                   const float* __restrict__ w,
                                   float* sOut, int ldso, int K) {
  const int b = blockIdx.z;
  const int m0 = blockIdx.y * 64, n0 = blockIdx.x * 32;
  v8f acc[8] = {};
  mm64x32(A + sA * b, lda, Bt + sB * b, ldb, K, m0, n0, acc);
  const int lane = threadIdx.x & 31, g = lane >> 4, i = lane & 15;
  const bf16_t* Ab = Add + sAdd * b;
#pragma unroll
  for (int nt = 0; nt < 2; ++nt) {
    const int n = n0 + nt * 16 + i;
    float partial = 0.f;
#pragma unroll
    for (int mt = 0; mt < 4; ++mt) {
      v8f a = acc[mt * 2 + nt];
#pragma unroll
      for (int j = 0; j < 8; ++j) {
        const int c = m0 + mt * 16 + 8 * g + j;
        float hv = tanhf(a[j] + (float)Ab[(size_t)c * ldadd + n]);
        partial += w[c] * hv;
      }
    }
    partial += __shfl_xor(partial, 16);     // combine rows j and j+8 (lane ^ 16)
    if (g == 0) atomicAdd(&sOut[(size_t)b * ldso + n], partial);
  }
}

// ---------------- data-prep kernels ----------------
__global__ void k_cast_bf16(const float* __restrict__ src, bf16_t* __restrict__ dst, size_t n) {
  size_t t = (size_t)blockIdx.x * blockDim.x + threadIdx.x;
  size_t stride = (size_t)gridDim.x * blockDim.x;
  for (; t < n; t += stride) dst[t] = (bf16_t)src[t];
}

// Vt[b][r][h] = bf16(V[b][h][r]); r in [0,RP_), zero-pad r >= R_
// Staged through LDS; uses CDNA5 async global->LDS DMA when available.
__global__ void k_transpose_V(const float* __restrict__ V, bf16_t* __restrict__ Vt) {
  __shared__ float tile[32][33];
  const int b = blockIdx.z;
  const int r0 = blockIdx.x * 32;
  const int h0 = blockIdx.y * 32;
  const int tx = threadIdx.x, ty = threadIdx.y;   // (32, 8)
#if __has_builtin(__builtin_amdgcn_global_load_async_to_lds_b32)
#pragma unroll
  for (int j = 0; j < 4; ++j) {
    const int h = h0 + ty + j * 8;
    const int r = r0 + tx;
    if (r < R_) {
      __builtin_amdgcn_global_load_async_to_lds_b32(
          (gint_p)(size_t)(V + ((size_t)b * H_ + h) * R_ + r),
          (lint_p)&tile[tx][ty + j * 8],
          0, 0);
    } else {
      tile[tx][ty + j * 8] = 0.f;
    }
  }
#if __has_builtin(__builtin_amdgcn_s_wait_asynccnt)
  __builtin_amdgcn_s_wait_asynccnt(0);
#else
  asm volatile("s_wait_asynccnt 0x0" ::: "memory");
#endif
#else
#pragma unroll
  for (int j = 0; j < 4; ++j) {
    const int h = h0 + ty + j * 8;
    const int r = r0 + tx;
    tile[tx][ty + j * 8] = (r < R_) ? V[((size_t)b * H_ + h) * R_ + r] : 0.f;
  }
#endif
  __syncthreads();
#pragma unroll
  for (int j = 0; j < 4; ++j) {
    const int r = r0 + ty + j * 8;
    const int h = h0 + tx;
    Vt[((size_t)b * RP_ + r) * H_ + h] = (bf16_t)tile[ty + j * 8][tx];
  }
}

// ---------------- softmax / gather epilogues (f32) ----------------
__device__ __forceinline__ float blk_reduce(float v, bool is_max) {
  __shared__ float sh[8];
  const int tid = threadIdx.x;
#pragma unroll
  for (int o = 16; o; o >>= 1) {
    float t = __shfl_xor(v, o);
    v = is_max ? fmaxf(v, t) : v + t;
  }
  if ((tid & 31) == 0) sh[tid >> 5] = v;
  __syncthreads();
  if (tid < 8) {
    v = sh[tid];
#pragma unroll
    for (int o = 4; o; o >>= 1) {
      float t = __shfl_xor(v, o);
      v = is_max ? fmaxf(v, t) : v + t;
    }
    if (tid == 0) sh[0] = v;
  }
  __syncthreads();
  float r = sh[0];
  __syncthreads();
  return r;
}

__global__ void k_softmax_av(const float* __restrict__ s_v, float* __restrict__ out_av) {
  const int b = blockIdx.x, tid = threadIdx.x;
  float x = (tid < R_) ? s_v[(size_t)b * RP_ + tid] : -1e30f;
  float m = blk_reduce(x, true);
  float e = (tid < R_) ? expf(x - m) : 0.f;
  float s = blk_reduce(e, false);
  if (tid < R_) out_av[(size_t)b * R_ + tid] = e / s;
}

// double softmax: a_q = softmax(s_q); then masked re-softmax of a_q
__global__ void k_softmax_aq(const float* __restrict__ s_q, const int* __restrict__ qlen,
                             float* __restrict__ out_aq) {
  const int b = blockIdx.x, tid = threadIdx.x;
  float x[4], p[4], e2[4];
  float mloc = -1e30f;
#pragma unroll
  for (int j = 0; j < 4; ++j) { x[j] = s_q[(size_t)b * L_ + tid + j * 256]; mloc = fmaxf(mloc, x[j]); }
  float m1 = blk_reduce(mloc, true);
  float sloc = 0.f;
#pragma unroll
  for (int j = 0; j < 4; ++j) { p[j] = expf(x[j] - m1); sloc += p[j]; }
  float s1 = blk_reduce(sloc, false);
  const int len = qlen[b];
  float m2loc = -1e30f;
#pragma unroll
  for (int j = 0; j < 4; ++j) {
    p[j] /= s1;
    if (tid + j * 256 < len) m2loc = fmaxf(m2loc, p[j]);
  }
  float m2 = blk_reduce(m2loc, true);
  float s2loc = 0.f;
#pragma unroll
  for (int j = 0; j < 4; ++j) {
    e2[j] = (tid + j * 256 < len) ? expf(p[j] - m2) : 0.f;
    s2loc += e2[j];
  }
  float s2 = blk_reduce(s2loc, false);
#pragma unroll
  for (int j = 0; j < 4; ++j) {
    const int l = tid + j * 256;
    out_aq[(size_t)b * L_ + l] = (l < len) ? e2[j] / s2 : 0.f;
  }
}

// v[b][h] = sum_r a_v[b][r] * V[b][h][r]   (f32 inputs)
__global__ void k_final_v(const float* __restrict__ V, const float* __restrict__ av,
                          float* __restrict__ out_v) {
  const int t = blockIdx.x * blockDim.x + threadIdx.x;
  if (t >= B_ * H_) return;
  const int b = t / H_, h = t % H_;
  const float* vp = V + ((size_t)b * H_ + h) * R_;
  const float* ap = av + (size_t)b * R_;
  float acc = 0.f;
  for (int r = 0; r < R_; ++r) acc += ap[r] * vp[r];
  out_v[t] = acc;
}

// q[b][h] = sum_l masked_a_q[b][l] * Q[b][l][h]
__global__ void k_final_q(const float* __restrict__ Q, const float* __restrict__ aq,
                          float* __restrict__ out_q) {
  const int b = blockIdx.x;
  const int h = blockIdx.y * 256 + threadIdx.x;
  const float* qp = Q + (size_t)b * L_ * H_ + h;
  const float* ap = aq + (size_t)b * L_;
  float acc = 0.f;
  for (int l = 0; l < L_; ++l) acc += ap[l] * qp[(size_t)l * H_];
  out_q[(size_t)b * H_ + h] = acc;
}

// =================================================================
extern "C" void kernel_launch(void* const* d_in, const int* in_sizes, int n_in,
                              void* d_out, int out_size, void* d_ws, size_t ws_size,
                              hipStream_t stream) {
  (void)in_sizes; (void)n_in; (void)out_size; (void)ws_size;
  const float* V    = (const float*)d_in[0];   // [B,H,R]
  const float* Q    = (const float*)d_in[1];   // [B,L,H]
  const int*   qlen = (const int*)  d_in[2];   // [B]
  const float* W_b  = (const float*)d_in[3];   // [H,H]
  const float* W_v  = (const float*)d_in[4];   // [CO,H]
  const float* W_q  = (const float*)d_in[5];   // [CO,H]
  const float* w_hv = (const float*)d_in[6];   // [CO,1]
  const float* w_hq = (const float*)d_in[7];   // [CO,1]

  float* out    = (float*)d_out;
  float* out_av = out;                              // B*R
  float* out_aq = out + (size_t)B_ * R_;            // B*L
  float* out_v  = out_aq + (size_t)B_ * L_;         // B*H
  float* out_q  = out_v + (size_t)B_ * H_;          // B*H

  // ---- workspace carve-up (~394 MB, bf16 intermediates) ----
  char* w = (char*)d_ws;
  auto alloc = [&](size_t bytes) -> char* {
    char* p = w; w += (bytes + 255) & ~(size_t)255; return p;
  };
  bf16_t* Qbf   = (bf16_t*)alloc((size_t)B_ * L_ * H_ * 2);    // [B][L][H]
  bf16_t* Vt    = (bf16_t*)alloc((size_t)B_ * RP_ * H_ * 2);   // [B][RP][H]  (V^T, zero-padded)
  bf16_t* Wbbf  = (bf16_t*)alloc((size_t)H_ * H_ * 2);         // [H][H]
  bf16_t* Wqbf  = (bf16_t*)alloc((size_t)CO_ * H_ * 2);        // [CO][H]
  bf16_t* Wvbf  = (bf16_t*)alloc((size_t)CO_ * H_ * 2);        // [CO][H]
  bf16_t* WbVt  = (bf16_t*)alloc((size_t)B_ * RP_ * H_ * 2);   // [B][RP][H]  ((W_b@V)^T)
  bf16_t* Cmat  = (bf16_t*)alloc((size_t)B_ * L_ * RP_ * 2);   // [B][L][RP]
  bf16_t* Ct    = (bf16_t*)alloc((size_t)B_ * RP_ * L_ * 2);   // [B][RP][L]
  bf16_t* WqQt  = (bf16_t*)alloc((size_t)B_ * CO_ * L_ * 2);   // [B][CO][L]
  bf16_t* WvV   = (bf16_t*)alloc((size_t)B_ * CO_ * RP_ * 2);  // [B][CO][RP]
  float*  s_v   = (float*) alloc((size_t)B_ * RP_ * 4);        // [B][RP]
  float*  s_q   = (float*) alloc((size_t)B_ * L_ * 4);         // [B][L]

  // ---- 1) precision conversion / layout ----
  k_cast_bf16<<<8192, 256, 0, stream>>>(Q, Qbf, (size_t)B_ * L_ * H_);
  k_cast_bf16<<<256, 256, 0, stream>>>(W_b, Wbbf, (size_t)H_ * H_);
  k_cast_bf16<<<128, 256, 0, stream>>>(W_q, Wqbf, (size_t)CO_ * H_);
  k_cast_bf16<<<128, 256, 0, stream>>>(W_v, Wvbf, (size_t)CO_ * H_);
  k_transpose_V<<<dim3(RP_ / 32, H_ / 32, B_), dim3(32, 8), 0, stream>>>(V, Vt);

  // ---- 2) WbVt[b] = (W_b @ V_b)^T : M=H,N=RP,K=H; A=W_b, Bt=Vt; store transposed ----
  k_gemm_storeT<<<dim3(RP_ / 32, H_ / 64, B_), 32, 0, stream>>>(
      Wbbf, H_, 0, Vt, H_, (size_t)RP_ * H_, WbVt, H_, (size_t)RP_ * H_, H_);

  // ---- 3) C[b] = Q_b @ WbV_b : M=L,N=RP,K=H; store both C and C^T ----
  k_gemm_storeBoth<<<dim3(RP_ / 32, L_ / 64, B_), 32, 0, stream>>>(
      Qbf, H_, (size_t)L_ * H_, WbVt, H_, (size_t)RP_ * H_,
      Cmat, RP_, (size_t)L_ * RP_, Ct, L_, (size_t)RP_ * L_, H_);

  // ---- 4) WqQt[b] = W_q @ Q_b^T : M=CO,N=L,K=H; Bt = Qbf (already [L][H]) ----
  k_gemm_storeN<<<dim3(L_ / 32, CO_ / 64, B_), 32, 0, stream>>>(
      Wqbf, H_, 0, Qbf, H_, (size_t)L_ * H_, WqQt, L_, (size_t)CO_ * L_, H_);

  // ---- 5) WvV[b] = W_v @ V_b : M=CO,N=RP,K=H; Bt = Vt ----
  k_gemm_storeN<<<dim3(RP_ / 32, CO_ / 64, B_), 32, 0, stream>>>(
      Wvbf, H_, 0, Vt, H_, (size_t)RP_ * H_, WvV, RP_, (size_t)CO_ * RP_, H_);

  (void)hipMemsetAsync(s_v, 0, (size_t)B_ * RP_ * 4, stream);
  (void)hipMemsetAsync(s_q, 0, (size_t)B_ * L_ * 4, stream);

  // ---- 6) s_v[b][r] = sum_c w_hv[c]*tanh(WvV + WqQt@C) : M=CO,N=RP,K=L ----
  k_gemm_tanh_reduce<<<dim3(RP_ / 32, CO_ / 64, B_), 32, 0, stream>>>(
      WqQt, L_, (size_t)CO_ * L_, Ct, L_, (size_t)RP_ * L_,
      WvV, RP_, (size_t)CO_ * RP_, w_hv, s_v, RP_, L_);

  // ---- 7) s_q[b][l] = sum_c w_hq[c]*tanh(WqQt + WvV@C^T) : M=CO,N=L,K=RP ----
  k_gemm_tanh_reduce<<<dim3(L_ / 32, CO_ / 64, B_), 32, 0, stream>>>(
      WvV, RP_, (size_t)CO_ * RP_, Cmat, RP_, (size_t)L_ * RP_,
      WqQt, L_, (size_t)CO_ * L_, w_hq, s_q, L_, RP_);

  // ---- 8) softmaxes + outputs ----
  k_softmax_av<<<B_, 256, 0, stream>>>(s_v, out_av);
  k_softmax_aq<<<B_, 256, 0, stream>>>(s_q, qlen, out_aq);
  k_final_v<<<(B_ * H_ + 255) / 256, 256, 0, stream>>>(V, out_av, out_v);
  k_final_q<<<dim3(B_, H_ / 256), 256, 0, stream>>>(Q, out_aq, out_q);
}
